// GraclusPooling_15006615733144
// MI455X (gfx1250) — compile-verified, matrix operand order checked by reading, not run on hardware
//
#include <hip/hip_runtime.h>
#include <hip/hip_bf16.h>

#define N_NODES 8192
#define DIM     128
#define K_POOL  4096
#define N_EDGES 8192

typedef __attribute__((ext_vector_type(16))) _Float16 v16h;
typedef __attribute__((ext_vector_type(2)))  _Float16 v2h;
typedef __attribute__((ext_vector_type(8)))  float    v8f;

// ---------------- workspace layout (bytes) ----------------
#define XN_OFF    0                         // _Float16 [N_NODES*DIM]  (2 MB)
#define DEG_OFF   (N_NODES * DIM * 2)       // float [N_NODES]
#define DIS_OFF   (DEG_OFF  + N_NODES * 4)  // float [N_NODES]
#define W_OFF     (DIS_OFF  + N_NODES * 4)  // float [N_NODES]
#define PERM_OFF  (W_OFF    + N_NODES * 4)  // int   [K_POOL]
#define MASK_OFF  (PERM_OFF + K_POOL  * 4)  // int   [N_NODES]
#define TOP_OFF   (MASK_OFF + N_NODES * 4)  // int   [N_NODES]
#define ZERO_N    ((TOP_OFF + N_NODES * 4 - DEG_OFF) / 4)

// ---------------- kernels ----------------

__global__ __launch_bounds__(256) void zero_ws_kernel(float* __restrict__ p, int n) {
    int i = blockIdx.x * 256 + threadIdx.x;
    if (i < n) p[i] = 0.0f;
}

// one wave (32 lanes) per row: L2-normalize, emit f16
__global__ __launch_bounds__(256) void normalize_kernel(const float* __restrict__ x,
                                                        _Float16* __restrict__ xn) {
    int wave = threadIdx.x >> 5;
    int lane = threadIdx.x & 31;
    int row  = blockIdx.x * 8 + wave;
    const float4 v = *(const float4*)(x + (size_t)row * DIM + lane * 4);
    float s = v.x * v.x + v.y * v.y + v.z * v.z + v.w * v.w;
    #pragma unroll
    for (int o = 16; o > 0; o >>= 1) s += __shfl_xor(s, o, 32);
    float inv = 1.0f / fmaxf(sqrtf(s), 1e-12f);
    _Float16* xo = xn + (size_t)row * DIM + lane * 4;
    xo[0] = (_Float16)(v.x * inv);
    xo[1] = (_Float16)(v.y * inv);
    xo[2] = (_Float16)(v.z * inv);
    xo[3] = (_Float16)(v.w * inv);
}

__global__ __launch_bounds__(256) void deg_kernel(const int* __restrict__ ei,
                                                  float* __restrict__ deg) {
    int e = blockIdx.x * 256 + threadIdx.x;
    if (e < N_EDGES) atomicAdd(&deg[ei[e]], 1.0f);
}

__global__ __launch_bounds__(256) void dis_kernel(const float* __restrict__ deg,
                                                  float* __restrict__ dis) {
    int i = blockIdx.x * 256 + threadIdx.x;
    if (i < N_NODES) {
        float d = deg[i];
        dis[i] = (d > 0.0f) ? rsqrtf(d) : 0.0f;
    }
}

// one wave per 16-edge group: A = xn[ei] rows (16x128), B = xn[ej] rows used as
// 128x16; four v_wmma_f32_16x16x32_f16 accumulate the 16x16 product whose
// diagonal is the per-edge cosine similarity.
__global__ __launch_bounds__(128) void edge_sim_kernel(const _Float16* __restrict__ xn,
                                                       const int* __restrict__ ei,
                                                       const int* __restrict__ ej,
                                                       const float* __restrict__ dis,
                                                       float* __restrict__ weights) {
    __shared__ __align__(16) _Float16 As[4][16 * DIM];
    __shared__ __align__(16) _Float16 Bs[4][16 * DIM];
    __shared__ float Ds[4][16 * 17];

    int wave = threadIdx.x >> 5;
    int lane = threadIdx.x & 31;
    int g = blockIdx.x * 4 + wave;              // 512 groups total
    const int* eg_i = ei + g * 16;
    const int* eg_j = ej + g * 16;
    __builtin_prefetch(eg_i, 0, 0);
    __builtin_prefetch(eg_j, 0, 0);

    // gather 16 rows (128 halves each) for A and B; 16B chunks, 8 per lane
    for (int c = lane; c < 256; c += 32) {
        int r  = c >> 4;
        int ch = c & 15;
        int ra = eg_i[r];
        int rb = eg_j[r];
        *(uint4*)&As[wave][r * DIM + ch * 8] =
            *(const uint4*)(xn + (size_t)ra * DIM + ch * 8);
        *(uint4*)&Bs[wave][r * DIM + ch * 8] =
            *(const uint4*)(xn + (size_t)rb * DIM + ch * 8);
    }
    __syncthreads();

    v8f acc = {};
    int m  = lane & 15;                 // A row / B column covered by this lane
    int kb = (lane < 16) ? 0 : 8;       // K sub-block base per ISA 16-bit layout
    #pragma unroll
    for (int kk = 0; kk < 4; ++kk) {    // K = 128 in 4 steps of 32
        v16h a, b;
        #pragma unroll
        for (int v = 0; v < 8; ++v) {
            int k0 = (v < 4) ? (kb + 2 * v) : (16 + kb + 2 * (v - 4));
            v2h pa = *(const v2h*)&As[wave][m * DIM + kk * 32 + k0];
            v2h pb = *(const v2h*)&Bs[wave][m * DIM + kk * 32 + k0];
            a[2 * v] = pa.x; a[2 * v + 1] = pa.y;
            b[2 * v] = pb.x; b[2 * v + 1] = pb.y;
        }
        acc = __builtin_amdgcn_wmma_f32_16x16x32_f16(
            false, a, false, b, (short)0, acc, false, false);
    }

    // D layout: lane covers column n = lane&15, VGPR v holds row v + 8*(lane>=16)
    int n     = lane & 15;
    int mbase = (lane < 16) ? 0 : 8;
    #pragma unroll
    for (int v = 0; v < 8; ++v) Ds[wave][(mbase + v) * 17 + n] = acc[v];
    __syncthreads();

    if (lane < 16) {
        float sim = Ds[wave][lane * 17 + lane];   // diagonal: dot(xn[ei], xn[ej])
        int i = eg_i[lane];
        int j = eg_j[lane];
        atomicAdd(&weights[i], dis[i] * sim * dis[j]);
    }
}

// single-WGP exact top-K: bitonic sort of 8192 (weight, idx) pairs in LDS,
// then perm/new_mask/keep-scan/edge-remap, all in one kernel.
__global__ __launch_bounds__(1024) void topk_kernel(const float* __restrict__ weights,
                                                    const int* __restrict__ ei,
                                                    const int* __restrict__ ej,
                                                    const int* __restrict__ batch,
                                                    int* __restrict__ permInt,
                                                    int* __restrict__ newMask,
                                                    int* __restrict__ isTop,
                                                    float* __restrict__ out) {
    __shared__ float key[N_NODES];
    __shared__ int   sidx[N_NODES];
    int tid = threadIdx.x;

    for (int i = tid; i < N_NODES; i += 1024) { key[i] = weights[i]; sidx[i] = i; }
    __syncthreads();

    // bitonic sort: descending by key, ascending index tie-break
    for (unsigned k = 2; k <= N_NODES; k <<= 1) {
        for (unsigned j = k >> 1; j > 0; j >>= 1) {
            for (unsigned i = tid; i < N_NODES; i += 1024) {
                unsigned p = i ^ j;
                if (p > i) {
                    bool  desc = ((i & k) == 0);
                    float ka = key[i], kp = key[p];
                    int   ia = sidx[i], ip = sidx[p];
                    bool aFirst = (ka > kp) || (ka == kp && ia < ip);
                    if (desc != aFirst) {
                        key[i] = kp; key[p] = ka;
                        sidx[i] = ip; sidx[p] = ia;
                    }
                }
            }
            __syncthreads();
        }
    }

    float* out_perm  = out + (size_t)K_POOL * DIM + 2 * (N_NODES - K_POOL);
    float* out_batch = out_perm + K_POOL;
    float* out_w     = out_batch + K_POOL;
    for (int r = tid; r < K_POOL; r += 1024) {
        int node = sidx[r];
        permInt[r]   = node;
        out_perm[r]  = (float)node;
        out_w[r]     = key[r];
        out_batch[r] = (float)batch[node];
        isTop[node]  = 1;
        newMask[node] = r;
    }
    __syncthreads();

    // prefix sum of non-top nodes (reuse sidx[0..1023] as scan buffer)
    int base = tid * 8;
    int cnt = 0;
    #pragma unroll
    for (int n = 0; n < 8; ++n) cnt += isTop[base + n] ? 0 : 1;
    sidx[tid] = cnt;
    __syncthreads();
    for (int off = 1; off < 1024; off <<= 1) {
        int v = (tid >= off) ? sidx[tid - off] : 0;
        __syncthreads();
        sidx[tid] += v;
        __syncthreads();
    }
    int pos = sidx[tid] - cnt;   // exclusive prefix

    float* out_ei = out + (size_t)K_POOL * DIM;
    float* out_ej = out_ei + (N_NODES - K_POOL);
    for (int n = base; n < base + 8; ++n) {
        if (!isTop[n]) {
            out_ei[pos] = (float)newMask[ei[n]];
            out_ej[pos] = (float)newMask[ej[n]];
            ++pos;
        }
    }
}

__global__ __launch_bounds__(256) void gather_kernel(const float* __restrict__ x,
                                                     const int* __restrict__ permInt,
                                                     float* __restrict__ out) {
    int gid = blockIdx.x * 256 + threadIdx.x;   // < K_POOL * DIM
    int r = gid >> 7;
    int c = gid & (DIM - 1);
    out[gid] = x[(size_t)permInt[r] * DIM + c];
}

// ---------------- launch ----------------
extern "C" void kernel_launch(void* const* d_in, const int* in_sizes, int n_in,
                              void* d_out, int out_size, void* d_ws, size_t ws_size,
                              hipStream_t stream) {
    const float* x     = (const float*)d_in[0];
    const int*   edge  = (const int*)d_in[1];   // [2][N_EDGES]
    const int*   batch = (const int*)d_in[2];
    const int*   ei = edge;
    const int*   ej = edge + N_EDGES;
    float* out = (float*)d_out;

    char* ws = (char*)d_ws;
    _Float16* xn      = (_Float16*)(ws + XN_OFF);
    float*    deg     = (float*)(ws + DEG_OFF);
    float*    dis     = (float*)(ws + DIS_OFF);
    float*    weights = (float*)(ws + W_OFF);
    int*      permInt = (int*)(ws + PERM_OFF);
    int*      newMask = (int*)(ws + MASK_OFF);
    int*      isTop   = (int*)(ws + TOP_OFF);

    zero_ws_kernel<<<(ZERO_N + 255) / 256, 256, 0, stream>>>(deg, ZERO_N);
    normalize_kernel<<<N_NODES / 8, 256, 0, stream>>>(x, xn);
    deg_kernel<<<N_EDGES / 256, 256, 0, stream>>>(ei, deg);
    dis_kernel<<<N_NODES / 256, 256, 0, stream>>>(deg, dis);
    edge_sim_kernel<<<(N_EDGES / 16) / 4, 128, 0, stream>>>(xn, ei, ej, dis, weights);
    topk_kernel<<<1, 1024, 0, stream>>>(weights, ei, ej, batch,
                                        permInt, newMask, isTop, out);
    gather_kernel<<<(K_POOL * DIM) / 256, 256, 0, stream>>>(x, permInt, out);
}
